// Decoder_71683004170249
// MI455X (gfx1250) — matrix operand order repaired and checked by reading, not compiled, not run on hardware
//
#include <hip/hip_runtime.h>
#include <hip/hip_bf16.h>
#include <math.h>

// ---- problem dims ----
#define B_   64
#define TX_  128
#define H_   1024
#define E_   512
#define KY_  32000

typedef __attribute__((ext_vector_type(16))) __bf16        bf16x16;
typedef __attribute__((ext_vector_type(8)))  float         v8f;
typedef __attribute__((ext_vector_type(8)))  unsigned int  u32x8;

// Pack two f32 into one dword of two bf16 (truncation) with a single v_perm_b32:
// result = { hi[31:16], lo[31:16] }
__device__ inline unsigned int pack_bf16(float lo, float hi) {
  return __builtin_amdgcn_perm(__builtin_bit_cast(unsigned int, hi),
                               __builtin_bit_cast(unsigned int, lo),
                               0x07060302u);
}

// A row made of two concatenated regions: index i reads p0[i] if i < split else p1[i]
// (p1 is pre-shifted so the same index works). Split points are multiples of 8,
// so an 8-float run never straddles regions.
struct RowPair {
  const float* p0;
  const float* p1;
  int split;
};

// A fragment: 16x32 bf16 (contiguous K order used here, same as B; not the ISA table order).
// lane 0-15: M=lane, K=0..15; lane 16-31: M=lane-16, K=16..31.
// Two contiguous 8-float runs -> 4x b128 loads + 8x v_perm.
__device__ inline bf16x16 load_a_frag(const RowPair& rp, int k0, int lane) {
  const int kh = (lane >> 4) & 1;
  u32x8 pk;
#pragma unroll
  for (int g = 0; g < 2; ++g) {
    const int kb = k0 + kh * 16 + g * 8;
    const float* p = (kb < rp.split) ? (rp.p0 + kb) : (rp.p1 + kb);
    const float4 q0 = *(const float4*)p;
    const float4 q1 = *(const float4*)(p + 4);
    pk[g * 4 + 0] = pack_bf16(q0.x, q0.y);
    pk[g * 4 + 1] = pack_bf16(q0.z, q0.w);
    pk[g * 4 + 2] = pack_bf16(q1.x, q1.y);
    pk[g * 4 + 3] = pack_bf16(q1.z, q1.w);
  }
  return __builtin_bit_cast(bf16x16, pk);
}

// B fragment: 32x16 bf16 from row-major weight W[n][k] (B[k][n] = W[n0+n][k0+k]).
// lane 0-15: N=lane, K=0..15; lane 16-31: N=lane-16, K=16..31 (16 contiguous K).
__device__ inline bf16x16 load_b_frag(const float* __restrict__ W, int ldk,
                                      int n0, int k0, int lane) {
  const int n  = lane & 15;
  const int kh = (lane >> 4) & 1;
  const float* p = W + (size_t)(n0 + n) * (size_t)ldk + (size_t)(k0 + kh * 16);
  u32x8 pk;
#pragma unroll
  for (int q = 0; q < 4; ++q) {
    const float4 v = *(const float4*)(p + 4 * q);
    pk[q * 2 + 0] = pack_bf16(v.x, v.y);
    pk[q * 2 + 1] = pack_bf16(v.z, v.w);
  }
  return __builtin_bit_cast(bf16x16, pk);
}

// 64-row GEMM body: one wave owns all 4 m-tiles (full B=64) x 4 n-tiles.
// Each B fragment is converted once and feeds 4 WMMAs; the weight matrix is
// streamed from memory exactly once across the whole grid (critical for the
// 393 MB lsm_W: 1 pass = ~17us at 23.3 TB/s instead of 4 passes).
template <int KTOT>
__device__ inline void gemm64_body(const RowPair rp[4], const float* __restrict__ W,
                                   int ldk, int n0base, int lane, v8f acc[4][4]) {
  for (int k0 = 0; k0 < KTOT; k0 += 32) {
    bf16x16 a[4];
#pragma unroll
    for (int mi = 0; mi < 4; ++mi) a[mi] = load_a_frag(rp[mi], k0, lane);
#pragma unroll
    for (int j = 0; j < 4; ++j) {
      bf16x16 bm = load_b_frag(W, ldk, n0base + j * 16, k0, lane);
#pragma unroll
      for (int mi = 0; mi < 4; ++mi)
        acc[mi][j] = __builtin_amdgcn_wmma_f32_16x16x32_bf16(
            false, a[mi], false, bm, (short)0, acc[mi][j], false, false);
    }
  }
}

// D/C tile (16x16 f32, 8 VGPRs): lane 0-15 -> N=lane, M=r; lane 16-31 -> N=lane-16, M=8+r.
__device__ inline void store_tiles64(float* __restrict__ out, int ldn, int n0base,
                                     const float* __restrict__ bias, int lane,
                                     v8f acc[4][4]) {
  const int n_in  = lane & 15;
  const int mbase = (lane >> 4) * 8;
#pragma unroll
  for (int mi = 0; mi < 4; ++mi) {
#pragma unroll
    for (int j = 0; j < 4; ++j) {
      const int n = n0base + j * 16 + n_in;
      const float bv = bias[n];
#pragma unroll
      for (int r2 = 0; r2 < 8; ++r2) {
        out[(size_t)(mi * 16 + mbase + r2) * (size_t)ldn + n] = acc[mi][j][r2] + bv;
      }
    }
  }
}

// ---------------------------------------------------------------------------
// K1: score[t*B+b] = aW2 . tanh([hidden(b) | enc(t,b)] @ aW1^T + ab1) + ab2
// One wave per 16 rows of the flattened (Tx*B, 3H) A matrix; the 16x1024 tanh
// activation never leaves registers. aW1 (12.6 MB) is L2-resident, so its
// per-block re-reads are cheap; enc (67 MB) also fits L2.
// ---------------------------------------------------------------------------
__global__ __launch_bounds__(32) void attn_score_kernel(
    const float* __restrict__ hidden, const float* __restrict__ enc,
    const float* __restrict__ aW1, const float* __restrict__ ab1,
    const float* __restrict__ aW2, const float* __restrict__ ab2,
    float* __restrict__ score) {
  const int lane = threadIdx.x;
  const int r0   = blockIdx.x * 16;
  const int r    = r0 + (lane & 15);
  const int b    = r & (B_ - 1);

  RowPair rp;
  rp.p0 = hidden + (size_t)b * H_;
  rp.p1 = enc + (size_t)r * (2 * H_) - H_;   // shifted: index i>=H_ hits enc row
  rp.split = H_;

  float partial[8];
#pragma unroll
  for (int i = 0; i < 8; ++i) partial[i] = 0.f;
  const int n_in = lane & 15;

  for (int ng = 0; ng < H_ / 64; ++ng) {         // 16 groups of 4 N-tiles
    v8f acc[4] = {};
    for (int k0 = 0; k0 < 3 * H_; k0 += 32) {    // 96 K steps
      bf16x16 a = load_a_frag(rp, k0, lane);
#pragma unroll
      for (int j = 0; j < 4; ++j) {
        bf16x16 bm = load_b_frag(aW1, 3 * H_, ng * 64 + j * 16, k0, lane);
        acc[j] = __builtin_amdgcn_wmma_f32_16x16x32_bf16(
            false, a, false, bm, (short)0, acc[j], false, false);
      }
    }
#pragma unroll
    for (int j = 0; j < 4; ++j) {
      const int nn = ng * 64 + j * 16 + n_in;
      const float w2 = aW2[nn];
      const float b1 = ab1[nn];
#pragma unroll
      for (int r2 = 0; r2 < 8; ++r2)
        partial[r2] += w2 * tanhf(acc[j][r2] + b1);
    }
  }
  // reduce over the 16 N-lanes within each half-wave
#pragma unroll
  for (int mask = 1; mask <= 8; mask <<= 1) {
#pragma unroll
    for (int r2 = 0; r2 < 8; ++r2)
      partial[r2] += __shfl_xor(partial[r2], mask, 32);
  }
  if ((lane & 15) == 0) {
    const int mbase = (lane >> 4) * 8;
    const float c = ab2[0];
#pragma unroll
    for (int r2 = 0; r2 < 8; ++r2)
      score[r0 + mbase + r2] = partial[r2] + c;
  }
}

// ---------------------------------------------------------------------------
// K2: per-b softmax over Tx, then context[b,:] = sum_t w[t] * enc[t,b,:]
// ---------------------------------------------------------------------------
__global__ void softmax_context_kernel(const float* __restrict__ score,
                                       const float* __restrict__ enc,
                                       float* __restrict__ ctx) {
  const int b = blockIdx.x, tid = threadIdx.x;
  __shared__ float sw[TX_];
  __shared__ float red[256];

  float s = (tid < TX_) ? score[(size_t)tid * B_ + b] : -3.4e38f;
  red[tid] = s;
  __syncthreads();
  for (int off = 128; off > 0; off >>= 1) {
    if (tid < off) red[tid] = fmaxf(red[tid], red[tid + off]);
    __syncthreads();
  }
  const float mx = red[0];
  __syncthreads();
  float e = (tid < TX_) ? __expf(s - mx) : 0.f;
  red[tid] = e;
  __syncthreads();
  for (int off = 128; off > 0; off >>= 1) {
    if (tid < off) red[tid] += red[tid + off];
    __syncthreads();
  }
  const float denom = red[0];
  if (tid < TX_) sw[tid] = e / denom;
  __syncthreads();

  for (int k = tid; k < 2 * H_; k += 256) {
    float acc = 0.f;
    for (int t = 0; t < TX_; ++t)
      acc += sw[t] * enc[((size_t)t * B_ + b) * (2 * H_) + k];
    ctx[(size_t)b * (2 * H_) + k] = acc;
  }
}

// ---------------------------------------------------------------------------
// K3: gi = [emb_W[input] | ctx] @ Wih^T + bih      (64x2560)x(2560x3072)
// ---------------------------------------------------------------------------
__global__ __launch_bounds__(32) void gi_kernel(
    const int* __restrict__ input, const float* __restrict__ embW,
    const float* __restrict__ ctx, const float* __restrict__ Wih,
    const float* __restrict__ bih, float* __restrict__ gi) {
  const int lane = threadIdx.x;
  const int n0base = blockIdx.x * 64;

  RowPair rp[4];
#pragma unroll
  for (int mi = 0; mi < 4; ++mi) {
    const int b = mi * 16 + (lane & 15);
    rp[mi].p0 = embW + (size_t)input[b] * E_;
    rp[mi].p1 = ctx + (size_t)b * (2 * H_) - E_;
    rp[mi].split = E_;
  }

  v8f acc[4][4] = {};
  gemm64_body<E_ + 2 * H_>(rp, Wih, E_ + 2 * H_, n0base, lane, acc);
  store_tiles64(gi, 3 * H_, n0base, bih, lane, acc);
}

// ---------------------------------------------------------------------------
// K4: gh = hidden @ Whh^T + bhh                    (64x1024)x(1024x3072)
// ---------------------------------------------------------------------------
__global__ __launch_bounds__(32) void gh_kernel(
    const float* __restrict__ hidden, const float* __restrict__ Whh,
    const float* __restrict__ bhh, float* __restrict__ gh) {
  const int lane = threadIdx.x;
  const int n0base = blockIdx.x * 64;

  RowPair rp[4];
#pragma unroll
  for (int mi = 0; mi < 4; ++mi) {
    const int b = mi * 16 + (lane & 15);
    rp[mi].p0 = hidden + (size_t)b * H_;
    rp[mi].p1 = rp[mi].p0;
    rp[mi].split = 1 << 30;
  }

  v8f acc[4][4] = {};
  gemm64_body<H_>(rp, Whh, H_, n0base, lane, acc);
  store_tiles64(gh, 3 * H_, n0base, bhh, lane, acc);
}

// ---------------------------------------------------------------------------
// K5: GRU gates -> h_new (written straight into d_out's second segment)
// ---------------------------------------------------------------------------
__global__ void gate_kernel(const float* __restrict__ gi, const float* __restrict__ gh,
                            const float* __restrict__ hidden, float* __restrict__ hnew) {
  const int idx = blockIdx.x * blockDim.x + threadIdx.x;
  if (idx >= B_ * H_) return;
  const int b = idx >> 10;
  const int h = idx & (H_ - 1);
  const size_t base = (size_t)b * (3 * H_);
  const float i_r = gi[base + h], i_z = gi[base + H_ + h], i_n = gi[base + 2 * H_ + h];
  const float h_r = gh[base + h], h_z = gh[base + H_ + h], h_n = gh[base + 2 * H_ + h];
  const float rr = 1.f / (1.f + __expf(-(i_r + h_r)));
  const float zz = 1.f / (1.f + __expf(-(i_z + h_z)));
  const float nn = tanhf(i_n + rr * h_n);
  hnew[idx] = (1.f - zz) * nn + zz * hidden[idx];
}

// ---------------------------------------------------------------------------
// K6: logits = [h_new | ctx] @ lsm_W^T + lsm_b     (64x3072)x(3072x32000)
// One pass over the 393 MB lsm_W -> ~17us bandwidth floor at 23.3 TB/s.
// ---------------------------------------------------------------------------
__global__ __launch_bounds__(32) void logits_kernel(
    const float* __restrict__ hnew, const float* __restrict__ ctx,
    const float* __restrict__ lsmW, const float* __restrict__ lsmb,
    float* __restrict__ logits) {
  const int lane = threadIdx.x;
  const int n0base = blockIdx.x * 64;

  RowPair rp[4];
#pragma unroll
  for (int mi = 0; mi < 4; ++mi) {
    const int b = mi * 16 + (lane & 15);
    rp[mi].p0 = hnew + (size_t)b * H_;
    rp[mi].p1 = ctx + (size_t)b * (2 * H_) - H_;
    rp[mi].split = H_;
  }

  v8f acc[4][4] = {};
  gemm64_body<3 * H_>(rp, lsmW, 3 * H_, n0base, lane, acc);
  store_tiles64(logits, KY_, n0base, lsmb, lane, acc);
}

// ---------------------------------------------------------------------------
// K7: log_softmax over Ky per row
// ---------------------------------------------------------------------------
__global__ void logsoftmax_kernel(const float* __restrict__ logits,
                                  float* __restrict__ out) {
  const int b = blockIdx.x, tid = threadIdx.x;
  __shared__ float red[256];
  const float* row = logits + (size_t)b * KY_;

  float mx = -3.4e38f;
  for (int k = tid; k < KY_; k += 256) mx = fmaxf(mx, row[k]);
  red[tid] = mx;
  __syncthreads();
  for (int off = 128; off > 0; off >>= 1) {
    if (tid < off) red[tid] = fmaxf(red[tid], red[tid + off]);
    __syncthreads();
  }
  mx = red[0];
  __syncthreads();

  float s = 0.f;
  for (int k = tid; k < KY_; k += 256) s += __expf(row[k] - mx);
  red[tid] = s;
  __syncthreads();
  for (int off = 128; off > 0; off >>= 1) {
    if (tid < off) red[tid] += red[tid + off];
    __syncthreads();
  }
  const float lse = mx + __logf(red[0]);

  for (int k = tid; k < KY_; k += 256)
    out[(size_t)b * KY_ + k] = row[k] - lse;
}

// ---------------------------------------------------------------------------
extern "C" void kernel_launch(void* const* d_in, const int* in_sizes, int n_in,
                              void* d_out, int out_size, void* d_ws, size_t ws_size,
                              hipStream_t stream) {
  (void)in_sizes; (void)n_in; (void)out_size; (void)ws_size;

  const int*   input  = (const int*)  d_in[0];
  const float* hidden = (const float*)d_in[1];
  const float* enc    = (const float*)d_in[2];
  const float* embW   = (const float*)d_in[3];
  const float* Wih    = (const float*)d_in[4];
  const float* Whh    = (const float*)d_in[5];
  const float* bih    = (const float*)d_in[6];
  const float* bhh    = (const float*)d_in[7];
  const float* lsmW   = (const float*)d_in[8];
  const float* lsmb   = (const float*)d_in[9];
  const float* aW1    = (const float*)d_in[10];
  const float* ab1    = (const float*)d_in[11];
  const float* aW2    = (const float*)d_in[12];
  const float* ab2    = (const float*)d_in[13];

  float* out      = (float*)d_out;
  float* hnew_out = out + (size_t)B_ * KY_;   // tuple: (output, h_new)

  // workspace partition (floats): total ~2.58M floats (~10.3 MB)
  float* ws     = (float*)d_ws;
  float* score  = ws;                              // Tx*B      = 8192
  float* ctx    = score + (size_t)TX_ * B_;        // B*2H      = 131072
  float* gi     = ctx   + (size_t)B_ * 2 * H_;     // B*3H      = 196608
  float* gh     = gi    + (size_t)B_ * 3 * H_;     // B*3H      = 196608
  float* logits = gh    + (size_t)B_ * 3 * H_;     // B*Ky      = 2048000

  attn_score_kernel<<<(TX_ * B_) / 16, 32, 0, stream>>>(hidden, enc, aW1, ab1, aW2, ab2, score);
  softmax_context_kernel<<<B_, 256, 0, stream>>>(score, enc, ctx);
  gi_kernel<<<(3 * H_) / 64, 32, 0, stream>>>(input, embW, ctx, Wih, bih, gi);
  gh_kernel<<<(3 * H_) / 64, 32, 0, stream>>>(hidden, Whh, bhh, gh);
  gate_kernel<<<(B_ * H_) / 256, 256, 0, stream>>>(gi, gh, hidden, hnew_out);
  logits_kernel<<<KY_ / 64, 32, 0, stream>>>(hnew_out, ctx, lsmW, lsmb, logits);
  logsoftmax_kernel<<<B_, 256, 0, stream>>>(logits, out);
}
